// LinearAttention_62130996904037
// MI455X (gfx1250) — matrix operand order repaired and checked by reading, not compiled, not run on hardware
//
#include <hip/hip_runtime.h>
#include <hip/hip_bf16.h>

// ---------------------------------------------------------------------------
// Model dims (fixed by the reference)
// ---------------------------------------------------------------------------
#define BB      2
#define TT      4096
#define HID     1024
#define NH      8
#define DK      128
#define DV      128
#define CHUNK   64
#define ROWS    (BB * TT)           // 8192
#define DVB     32                  // DV split block for the attention scan

typedef __attribute__((ext_vector_type(16))) _Float16 v16h;
typedef __attribute__((ext_vector_type(8)))  float    v8f;

// ---------------------------------------------------------------------------
// CDNA5 async global->LDS copies (ASYNCcnt-tracked; VGPR-bypassing staging)
// dst-first syntax per VGLOBAL encoding: VDST = LDS byte address VGPR.
// ---------------------------------------------------------------------------
__device__ __forceinline__ void async_ld128(const void* gaddr, void* lds) {
  unsigned l = (unsigned)(size_t)lds;
  asm volatile("global_load_async_to_lds_b128 %0, %1, off"
               :: "v"(l), "v"(gaddr) : "memory");
}
__device__ __forceinline__ void async_ld64(const void* gaddr, void* lds) {
  unsigned l = (unsigned)(size_t)lds;
  asm volatile("global_load_async_to_lds_b64 %0, %1, off"
               :: "v"(l), "v"(gaddr) : "memory");
}
__device__ __forceinline__ void wait_async_2() {
  asm volatile("s_wait_asynccnt 0x2" ::: "memory");
}
__device__ __forceinline__ void wait_async_0() {
  asm volatile("s_wait_asynccnt 0x0" ::: "memory");
}

// ---------------------------------------------------------------------------
// WMMA fragment helpers (wave32, 16x16x32 f16 -> f32), A/B element s at k = 16*(lane/16) + s
// ---------------------------------------------------------------------------
__device__ __forceinline__ v16h load_a_frag(const _Float16* __restrict__ src,
                                            int ld, int row0, int col0, int lane) {
  int m  = lane & 15;
  int kb = (lane < 16) ? 0 : 16;
  const _Float16* p = src + (long)(row0 + m) * ld + col0;
  v16h a;
#pragma unroll
  for (int r = 0; r < 8; ++r) {
    int k0 = kb + 2 * r;
    a[2 * r]     = p[k0];
    a[2 * r + 1] = p[k0 + 1];
  }
  return a;
}

// A[m,k] = src[(col0+k)*ld + row0 + m]   (transposed source, e.g. k^T)
__device__ __forceinline__ v16h load_a_frag_trans(const _Float16* __restrict__ src,
                                                  int ld, int row0, int col0, int lane) {
  int m  = lane & 15;
  int kb = (lane < 16) ? 0 : 16;
  const _Float16* p = src + (long)col0 * ld + row0 + m;
  v16h a;
#pragma unroll
  for (int r = 0; r < 8; ++r) {
    int k0 = kb + 2 * r;
    a[2 * r]     = p[(long)k0 * ld];
    a[2 * r + 1] = p[(long)(k0 + 1) * ld];
  }
  return a;
}

// B[k,n] = src[(k0+k)*ld + n0 + n]  (k-major source: V tiles, state S rows)
__device__ __forceinline__ v16h load_b_frag_kmaj(const _Float16* __restrict__ src,
                                                 int ld, int k0, int n0, int lane) {
  int n  = lane & 15;
  int kb = (lane < 16) ? 0 : 16;
  const _Float16* p = src + (long)(k0 + kb) * ld + n0 + n;
  v16h b;
#pragma unroll
  for (int r = 0; r < 8; ++r) {
    b[2 * r]     = p[(long)(2 * r) * ld];
    b[2 * r + 1] = p[(long)(2 * r + 1) * ld];
  }
  return b;
}

// same, f32 source (state S) with on-the-fly f16 conversion
__device__ __forceinline__ v16h load_b_frag_kmaj_f32(const float* __restrict__ src,
                                                     int ld, int k0, int n0, int lane) {
  int n  = lane & 15;
  int kb = (lane < 16) ? 0 : 16;
  const float* p = src + (long)(k0 + kb) * ld + n0 + n;
  v16h b;
#pragma unroll
  for (int r = 0; r < 8; ++r) {
    b[2 * r]     = (_Float16)p[(long)(2 * r) * ld];
    b[2 * r + 1] = (_Float16)p[(long)(2 * r + 1) * ld];
  }
  return b;
}

// B[k,n] = src[(n0+n)*ld + k0 + k]  (n-major source: weight rows / k-chunk)
__device__ __forceinline__ v16h load_b_frag_nmaj(const _Float16* __restrict__ src,
                                                 int ld, int k0, int n0, int lane) {
  int n  = lane & 15;
  int kb = (lane < 16) ? 0 : 16;
  const _Float16* p = src + (long)(n0 + n) * ld + k0 + kb;
  v16h b;
#pragma unroll
  for (int r = 0; r < 8; ++r) {
    b[2 * r]     = p[2 * r];
    b[2 * r + 1] = p[2 * r + 1];
  }
  return b;
}

__device__ __forceinline__ void store_c_frag(float* __restrict__ dst, int ld,
                                             int row0, int col0, int lane, v8f c) {
  int n  = lane & 15;
  int mb = (lane < 16) ? 0 : 8;
  float* p = dst + (long)(row0 + mb) * ld + col0 + n;
#pragma unroll
  for (int r = 0; r < 8; ++r) p[(long)r * ld] = c[r];
}

__device__ __forceinline__ v8f load_c_frag(const float* __restrict__ src, int ld,
                                           int row0, int col0, int lane) {
  int n  = lane & 15;
  int mb = (lane < 16) ? 0 : 8;
  const float* p = src + (long)(row0 + mb) * ld + col0 + n;
  v8f c;
#pragma unroll
  for (int r = 0; r < 8; ++r) c[r] = p[(long)r * ld];
  return c;
}

#define WMMA_F16(a, b, c) \
  __builtin_amdgcn_wmma_f32_16x16x32_f16(false, (a), false, (b), (short)0, (c), false, false)

// ---------------------------------------------------------------------------
// Tiled GEMM: C[M,N] = A[M,K] @ B[N,K]^T  (row-major, f16 in, f32 out)
// 128x128 macro tile, 16 waves; each wave owns a 2x2 grid of 16x16 WMMA tiles.
// K staged in 32-wide slabs through double-buffered LDS fed by
// GLOBAL_LOAD_ASYNC_TO_LDS_B128 (issued one slab ahead). grid.z = batch.
// ---------------------------------------------------------------------------
#define GT_M 128
#define GT_N 128
#define GT_K 32

__global__ void __launch_bounds__(512)
gemm_f16_wmma(const _Float16* __restrict__ A, int lda, long aBatch,
              const _Float16* __restrict__ Bw, int ldb, long bBatch,
              float* __restrict__ C, int ldc, long cBatch, int K) {
  __shared__ _Float16 As[2][GT_M][GT_K];   // 2 x 8 KB
  __shared__ _Float16 Bs[2][GT_N][GT_K];   // 2 x 8 KB

  long z = blockIdx.z;
  A  += z * aBatch;
  Bw += z * bBatch;
  C  += z * cBatch;

  const int m0   = blockIdx.y * GT_M;
  const int n0   = blockIdx.x * GT_N;
  const int tid  = threadIdx.x;
  const int lane = tid & 31;
  const int wave = tid >> 5;
  const int wr   = wave >> 2;   // 0..3 -> 32-row band
  const int wc   = wave & 3;    // 0..3 -> 32-col band

  // staging split: 512 threads x 16B = 8 KB per matrix per slab
  const int sidx = tid * 8;            // half index within 128x32 slab
  const int sr   = sidx >> 5;          // 0..127
  const int sc   = sidx & 31;          // 0,8,16,24

  const int nslab = K / GT_K;

  // prologue: stage slab 0 into buffer 0
  async_ld128(A  + (long)(m0 + sr) * lda + sc, &As[0][sr][sc]);
  async_ld128(Bw + (long)(n0 + sr) * ldb + sc, &Bs[0][sr][sc]);

  v8f acc00 = {}, acc01 = {}, acc10 = {}, acc11 = {};

  for (int s = 0; s < nslab; ++s) {
    const int cur = s & 1;
    if (s + 1 < nslab) {
      const int nxt = (s + 1) & 1;
      const long kn = (long)(s + 1) * GT_K;
      async_ld128(A  + (long)(m0 + sr) * lda + kn + sc, &As[nxt][sr][sc]);
      async_ld128(Bw + (long)(n0 + sr) * ldb + kn + sc, &Bs[nxt][sr][sc]);
      wait_async_2();   // own slab-s copies complete (2 newer in flight)
    } else {
      wait_async_0();
    }
    __syncthreads();    // all waves' slab-s staging visible

    v16h a0 = load_a_frag(&As[cur][0][0], GT_K, wr * 32,      0, lane);
    v16h a1 = load_a_frag(&As[cur][0][0], GT_K, wr * 32 + 16, 0, lane);
    v16h b0 = load_b_frag_nmaj(&Bs[cur][0][0], GT_K, 0, wc * 32,      lane);
    v16h b1 = load_b_frag_nmaj(&Bs[cur][0][0], GT_K, 0, wc * 32 + 16, lane);
    acc00 = WMMA_F16(a0, b0, acc00);
    acc01 = WMMA_F16(a0, b1, acc01);
    acc10 = WMMA_F16(a1, b0, acc10);
    acc11 = WMMA_F16(a1, b1, acc11);
    __syncthreads();    // done reading before buffer reuse next iteration
  }

  const int rm = m0 + wr * 32, cn = n0 + wc * 32;
  store_c_frag(C, ldc, rm,      cn,      lane, acc00);
  store_c_frag(C, ldc, rm,      cn + 16, lane, acc01);
  store_c_frag(C, ldc, rm + 16, cn,      lane, acc10);
  store_c_frag(C, ldc, rm + 16, cn + 16, lane, acc11);
}

// ---------------------------------------------------------------------------
// fp32 -> f16 conversion
// ---------------------------------------------------------------------------
__global__ void cvt_f32_to_f16(const float* __restrict__ in,
                               _Float16* __restrict__ out, long n) {
  long i = (long)blockIdx.x * blockDim.x + threadIdx.x;
  if (i < n) out[i] = (_Float16)in[i];
}

// ---------------------------------------------------------------------------
// causal depthwise conv (K=4) + SiLU; f32 in, f16 out
// ---------------------------------------------------------------------------
__global__ void conv_silu_kernel(const float* __restrict__ x,
                                 const float* __restrict__ w,
                                 _Float16* __restrict__ out) {
  long i = (long)blockIdx.x * blockDim.x + threadIdx.x;
  const long total = (long)ROWS * HID;
  if (i >= total) return;
  int d  = (int)(i % HID);
  long bt = i / HID;
  int t  = (int)(bt % TT);
  long b = bt / TT;
  float acc = 0.f;
#pragma unroll
  for (int j = 0; j < 4; ++j) {
    int ts = t - 3 + j;
    if (ts >= 0) acc += x[(b * TT + ts) * HID + d] * w[d * 4 + j];
  }
  float s = acc / (1.f + __expf(-acc));
  out[i] = (_Float16)s;
}

// ---------------------------------------------------------------------------
// Hadamard feature-map combine: out = (t1 + b1[e]) * (t2 + b2[e]),  e = d % DK
// ---------------------------------------------------------------------------
__global__ void fm_combine_kernel(const float* __restrict__ t1,
                                  const float* __restrict__ t2,
                                  const float* __restrict__ b1,
                                  const float* __restrict__ b2,
                                  _Float16* __restrict__ out) {
  long i = (long)blockIdx.x * blockDim.x + threadIdx.x;
  const long total = (long)ROWS * HID;
  if (i >= total) return;
  int e = (int)(i % HID) & (DK - 1);
  out[i] = (_Float16)((t1[i] + b1[e]) * (t2[i] + b2[e]));
}

// ---------------------------------------------------------------------------
// Chunked causal linear attention, DV split into 4 blocks of 32 columns.
// One workgroup per (b, h, dv-block): 64 WGs, 512 threads (16 waves).
// Per-WG state S[128][32] f32 held in LDS for the whole scan. Per chunk:
//   A: attn = tril(q k^T) * scale     (64x64, K=128; 16 waves, 4x4 tiles)
//   B: o = scale*(q @ S) + attn @ v   (64x32;  waves 0..7, 4x2 tiles)
//   C: S += k^T @ v                   (128x32; 16 waves, 8x2 tiles)
// q/k/v chunks staged raw via async-to-LDS (scale folded into A/B epilogues).
// ---------------------------------------------------------------------------
__global__ void __launch_bounds__(512)
chunk_attn_kernel(const _Float16* __restrict__ Qg,
                  const _Float16* __restrict__ Kg,
                  const _Float16* __restrict__ Vg,
                  float* __restrict__ Og, float scale) {
  __shared__ float    S [DK * DVB];        // 16 KB
  __shared__ _Float16 Qs[CHUNK * DK];      // 16 KB
  __shared__ _Float16 Ks[CHUNK * DK];      // 16 KB
  __shared__ _Float16 Vs[CHUNK * DVB];     //  4 KB
  __shared__ _Float16 As[CHUNK * CHUNK];   //  8 KB

  const int wg   = blockIdx.x;             // (b*NH + h)*4 + cb
  const int cb   = wg & 3;
  const int bh   = wg >> 2;
  const int b    = bh >> 3;
  const int h    = bh & 7;
  const int cv0  = cb * DVB;
  const int tid  = threadIdx.x;
  const int lane = tid & 31;
  const int wave = tid >> 5;

  const long rs = HID;
  const _Float16* Qb = Qg + ((long)b * TT) * rs + (long)h * DK;
  const _Float16* Kb = Kg + ((long)b * TT) * rs + (long)h * DK;
  const _Float16* Vb = Vg + ((long)b * TT) * rs + (long)h * DV + cv0;
  float*          Ob = Og + ((long)b * TT) * rs + (long)h * DV + cv0;

  for (int i = tid; i < DK * DVB; i += 512) S[i] = 0.f;
  __syncthreads();

  const int nc = TT / CHUNK;
  for (int c = 0; c < nc; ++c) {
    const long co = (long)c * CHUNK * rs;
    // ---- async staging of raw q, k (64x128) and v (64x32) chunks ----
    {
      const int qi = tid * 16;                   // 2 x b128 per thread
      const int qr = qi >> 7, qc = qi & 127;
      async_ld128(Qb + co + (long)qr * rs + qc,     &Qs[qi]);
      async_ld128(Qb + co + (long)qr * rs + qc + 8, &Qs[qi + 8]);
      async_ld128(Kb + co + (long)qr * rs + qc,     &Ks[qi]);
      async_ld128(Kb + co + (long)qr * rs + qc + 8, &Ks[qi + 8]);
      const int vi = tid * 4;                    // 1 x b64 per thread
      const int vr = vi >> 5, vc = vi & 31;
      async_ld64(Vb + co + (long)vr * rs + vc, &Vs[vi]);
    }
    wait_async_0();
    __syncthreads();

    // ---- Step A: attn = tril(scale * q k^T), 4x4 tiles, one per wave ----
    {
      const int tr = wave >> 2, tc = wave & 3;
      v8f acc = {};
#pragma unroll
      for (int kk = 0; kk < DK; kk += 32) {
        v16h a  = load_a_frag(Qs, DK, tr * 16, kk, lane);
        v16h bf = load_b_frag_nmaj(Ks, DK, kk, tc * 16, lane);
        acc = WMMA_F16(a, bf, acc);
      }
      const int n = lane & 15, mb = (lane < 16) ? 0 : 8;
#pragma unroll
      for (int r = 0; r < 8; ++r) {
        int row = tr * 16 + mb + r, col = tc * 16 + n;
        As[row * CHUNK + col] = (_Float16)((col <= row) ? acc[r] * scale : 0.f);
      }
    }
    __syncthreads();

    // ---- Step B: o = scale*(q @ S) + attn @ v  (4x2 tiles; waves 0..7) ----
    if (wave < 8) {
      const int tr = wave >> 1, tc = wave & 1;
      v8f acc = {};
#pragma unroll
      for (int kk = 0; kk < DK; kk += 32) {            // inter-chunk: q @ S
        v16h a  = load_a_frag(Qs, DK, tr * 16, kk, lane);
        v16h bf = load_b_frag_kmaj_f32(S, DVB, kk, tc * 16, lane);
        acc = WMMA_F16(a, bf, acc);
      }
#pragma unroll
      for (int r = 0; r < 8; ++r) acc[r] *= scale;     // fold q-scale
#pragma unroll
      for (int kk = 0; kk < CHUNK; kk += 32) {         // intra-chunk: attn @ v
        v16h a  = load_a_frag(As, CHUNK, tr * 16, kk, lane);
        v16h bf = load_b_frag_kmaj(Vs, DVB, kk, tc * 16, lane);
        acc = WMMA_F16(a, bf, acc);
      }
      store_c_frag(Ob + co, HID, tr * 16, tc * 16, lane, acc);
    }
    __syncthreads();   // all reads of S done before updating it

    // ---- Step C: S += k^T @ v  (8x2 tiles, one per wave) ----
    {
      const int tr = wave >> 1, tc = wave & 1;
      v8f acc = load_c_frag(S, DVB, tr * 16, tc * 16, lane);
#pragma unroll
      for (int kk = 0; kk < CHUNK; kk += 32) {
        v16h a  = load_a_frag_trans(Ks, DK, tr * 16, kk, lane);  // k^T
        v16h bf = load_b_frag_kmaj(Vs, DVB, kk, tc * 16, lane);
        acc = WMMA_F16(a, bf, acc);
      }
      store_c_frag(S, DVB, tr * 16, tc * 16, lane, acc);
    }
    __syncthreads();   // S updated, k/v reads done, before next staging
  }
}

// ---------------------------------------------------------------------------
// Host-side orchestration
// ---------------------------------------------------------------------------
static inline void* ws_take(char*& p, size_t bytes) {
  void* r = (void*)p;
  p += (bytes + 255) & ~(size_t)255;
  return r;
}

extern "C" void kernel_launch(void* const* d_in, const int* in_sizes, int n_in,
                              void* d_out, int out_size, void* d_ws, size_t ws_size,
                              hipStream_t stream) {
  (void)in_sizes; (void)n_in; (void)out_size; (void)ws_size;

  const float* x      = (const float*)d_in[0];
  const float* wq     = (const float*)d_in[1];
  const float* wk     = (const float*)d_in[2];
  const float* wv     = (const float*)d_in[3];
  const float* wo     = (const float*)d_in[4];
  const float* conv_q = (const float*)d_in[5];
  const float* conv_k = (const float*)d_in[6];
  const float* conv_v = (const float*)d_in[7];
  const float* fmq_w1 = (const float*)d_in[8];
  const float* fmq_b1 = (const float*)d_in[9];
  const float* fmq_w2 = (const float*)d_in[10];
  const float* fmq_b2 = (const float*)d_in[11];
  const float* fmk_w1 = (const float*)d_in[12];
  const float* fmk_b1 = (const float*)d_in[13];
  const float* fmk_w2 = (const float*)d_in[14];
  const float* fmk_b2 = (const float*)d_in[15];
  float* out = (float*)d_out;

  char* wp = (char*)d_ws;
  const long NBIG = (long)ROWS * HID;                 // 8.4M elements

  _Float16* xh    = (_Float16*)ws_take(wp, NBIG * 2);
  _Float16* wqh   = (_Float16*)ws_take(wp, (long)HID * HID * 2);
  _Float16* wkh   = (_Float16*)ws_take(wp, (long)HID * HID * 2);
  _Float16* wvh   = (_Float16*)ws_take(wp, (long)HID * HID * 2);
  _Float16* woh   = (_Float16*)ws_take(wp, (long)HID * HID * 2);
  _Float16* fq1h  = (_Float16*)ws_take(wp, (long)DK * DK * 2);
  _Float16* fq2h  = (_Float16*)ws_take(wp, (long)DK * DK * 2);
  _Float16* fk1h  = (_Float16*)ws_take(wp, (long)DK * DK * 2);
  _Float16* fk2h  = (_Float16*)ws_take(wp, (long)DK * DK * 2);
  float*    proj  = (float*)   ws_take(wp, NBIG * 4);
  _Float16* qcv   = (_Float16*)ws_take(wp, NBIG * 2);
  _Float16* kcv   = (_Float16*)ws_take(wp, NBIG * 2);
  _Float16* vcv   = (_Float16*)ws_take(wp, NBIG * 2);
  float*    t1    = (float*)   ws_take(wp, NBIG * 4);
  float*    t2    = (float*)   ws_take(wp, NBIG * 4);
  _Float16* qf    = (_Float16*)ws_take(wp, NBIG * 2);
  _Float16* kf    = (_Float16*)ws_take(wp, NBIG * 2);
  float*    obuf  = (float*)   ws_take(wp, NBIG * 4);
  _Float16* oh    = (_Float16*)ws_take(wp, NBIG * 2);

  const int EW_B = 256;
  auto cvt = [&](const float* src, _Float16* dst, long n) {
    cvt_f32_to_f16<<<dim3((unsigned)((n + EW_B - 1) / EW_B)), dim3(EW_B), 0, stream>>>(src, dst, n);
  };

  // f16 copies of activations and all weights
  cvt(x, xh, NBIG);
  cvt(wq, wqh, (long)HID * HID);
  cvt(wk, wkh, (long)HID * HID);
  cvt(wv, wvh, (long)HID * HID);
  cvt(wo, woh, (long)HID * HID);
  cvt(fmq_w1, fq1h, (long)DK * DK);
  cvt(fmq_w2, fq2h, (long)DK * DK);
  cvt(fmk_w1, fk1h, (long)DK * DK);
  cvt(fmk_w2, fk2h, (long)DK * DK);

  const dim3 gemmBlk(512);
  const dim3 gemmGridBig(HID / GT_N, ROWS / GT_M, 1);     // (8, 64)
  const dim3 ewGrid((unsigned)((NBIG + EW_B - 1) / EW_B));

  // ---- projections + conv/SiLU (proj buffer reused sequentially) ----
  gemm_f16_wmma<<<gemmGridBig, gemmBlk, 0, stream>>>(xh, HID, 0, wqh, HID, 0,
                                                     proj, HID, 0, HID);
  conv_silu_kernel<<<ewGrid, dim3(EW_B), 0, stream>>>(proj, conv_q, qcv);

  gemm_f16_wmma<<<gemmGridBig, gemmBlk, 0, stream>>>(xh, HID, 0, wkh, HID, 0,
                                                     proj, HID, 0, HID);
  conv_silu_kernel<<<ewGrid, dim3(EW_B), 0, stream>>>(proj, conv_k, kcv);

  gemm_f16_wmma<<<gemmGridBig, gemmBlk, 0, stream>>>(xh, HID, 0, wvh, HID, 0,
                                                     proj, HID, 0, HID);
  conv_silu_kernel<<<ewGrid, dim3(EW_B), 0, stream>>>(proj, conv_v, vcv);

  // ---- per-head feature maps: heads batched via grid.z ----
  const dim3 fmGrid(DK / GT_N, ROWS / GT_M, NH);          // (1, 64, 8)
  gemm_f16_wmma<<<fmGrid, gemmBlk, 0, stream>>>(qcv, HID, DK, fq1h, DK, 0,
                                                t1, HID, DK, DK);
  gemm_f16_wmma<<<fmGrid, gemmBlk, 0, stream>>>(qcv, HID, DK, fq2h, DK, 0,
                                                t2, HID, DK, DK);
  fm_combine_kernel<<<ewGrid, dim3(EW_B), 0, stream>>>(t1, t2, fmq_b1, fmq_b2, qf);

  gemm_f16_wmma<<<fmGrid, gemmBlk, 0, stream>>>(kcv, HID, DK, fk1h, DK, 0,
                                                t1, HID, DK, DK);
  gemm_f16_wmma<<<fmGrid, gemmBlk, 0, stream>>>(kcv, HID, DK, fk2h, DK, 0,
                                                t2, HID, DK, DK);
  fm_combine_kernel<<<ewGrid, dim3(EW_B), 0, stream>>>(t1, t2, fmk_b1, fmk_b2, kf);

  // ---- chunked linear attention scan: one WG per (b, h, dv-block) ----
  const float scale = 0.08838834764831845f;   // DK^-0.5
  chunk_attn_kernel<<<dim3(BB * NH * (DV / DVB)), dim3(512), 0, stream>>>(
      qf, kf, vcv, obuf, scale);

  // ---- output projection ----
  cvt(obuf, oh, NBIG);
  gemm_f16_wmma<<<gemmGridBig, gemmBlk, 0, stream>>>(oh, HID, 0, woh, HID, 0,
                                                     out, HID, 0, HID);
}
